// Decoder_62139586838916
// MI455X (gfx1250) — matrix operand order repaired and checked by reading, not compiled, not run on hardware
//
#include <hip/hip_runtime.h>
#include <hip/hip_bf16.h>
#include <stdint.h>

// ---------------------------------------------------------------------------
// CDNA5 (gfx1250) decoder: PreNet + 3x LSTM(768) + proj, B=16, T=1000.
//  - dense projections: WMMA f16 GEMMs (f32 accumulate), reg-double-buffered
//  - recurrent part: persistent 24-WG scan, w_hh slice resident in LDS
//    (192KB/WGP via TDM tensor_load_to_lds), grid barrier per timestep,
//    h broadcast via global_load_async_to_lds_b128 (ASYNCcnt).
// ---------------------------------------------------------------------------

typedef __attribute__((ext_vector_type(16))) _Float16 v16h;
typedef __attribute__((ext_vector_type(8)))  _Float16 v8h;
typedef __attribute__((ext_vector_type(8)))  float    v8f;
typedef __attribute__((ext_vector_type(4)))  unsigned int u32x4;
typedef __attribute__((ext_vector_type(8)))  int i32x8;
typedef __attribute__((ext_vector_type(4)))  int i32x4;

// ---- CDNA5 async global->LDS copy (ASYNCcnt) -------------------------------
static __device__ __forceinline__ void async_g2l_b128(void* lds, const void* g) {
  asm volatile("global_load_async_to_lds_b128 %0, %1, off"
               :: "v"((uint32_t)(uintptr_t)lds), "v"((uint64_t)(uintptr_t)g)
               : "memory");
}
static __device__ __forceinline__ void wait_async0() {
#if __has_builtin(__builtin_amdgcn_s_wait_asynccnt)
  __builtin_amdgcn_s_wait_asynccnt(0);
#else
  asm volatile("s_wait_asynccnt 0x0" ::: "memory");
#endif
}
static __device__ __forceinline__ void wait_tensor0() {
#if __has_builtin(__builtin_amdgcn_s_wait_tensorcnt)
  __builtin_amdgcn_s_wait_tensorcnt(0);
#else
  asm volatile("s_wait_tensorcnt 0x0" ::: "memory");
#endif
}

// ---- WMMA fragment loaders (layouts per CDNA5 ISA §7.12.2) ------
// A tile 16x32 (MxK) f16: lane l (0..15) holds row M=l, K=0..7 & 16..23;
// lane l+16 holds row M=l, K=8..15 & 24..31.
static __device__ __forceinline__ v16h loadA_frag(const _Float16* base, int ldh, int lane) {
  const int m  = lane & 15;
  const int h8 = (lane >> 4) * 8;
  const _Float16* p = base + (size_t)m * ldh + h8;
  union { v16h v; v8h h[2]; } u;
  u.h[0] = *(const v8h*)(p);
  u.h[1] = *(const v8h*)(p + 16);
  return u.v;
}
// B tile 32x16 (KxN) f16, packed N-major [16][32]: lane l holds col N=l%16,
// K=0..7 & 16..23 (lanes 0-15) or K=8..15 & 24..31 (lanes 16-31).
static __device__ __forceinline__ v16h loadB_frag(const _Float16* tile, int lane) {
  const _Float16* p = tile + (lane & 15) * 32 + (lane >> 4) * 8;
  union { v16h v; v8h h[2]; } u;
  u.h[0] = *(const v8h*)(p);
  u.h[1] = *(const v8h*)(p + 16);
  return u.v;
}
static __device__ __forceinline__ v8f wmma_f16(v16h a, v16h b, v8f c) {
  return __builtin_amdgcn_wmma_f32_16x16x32_f16(false, a, false, b, (short)0, c, false, false);
}

// ---- weight packer: W[N][K] f32 row-major -> f16 tiles (nt,kt)[16][32] ------
__global__ __launch_bounds__(256) void pack_w(const float* __restrict__ W,
                                              _Float16* __restrict__ out,
                                              int N, int K) {
  const int Kt = K >> 5;
  size_t e = (size_t)blockIdx.x * 256 + threadIdx.x;
  size_t tot = (size_t)N * K;
  if (e >= tot) return;
  size_t tile = e >> 9;
  int within = (int)(e & 511);
  int kk = within & 31, nn = within >> 5;
  int nt = (int)(tile / Kt), kt = (int)(tile % Kt);
  out[e] = (_Float16)W[(size_t)(nt * 16 + nn) * K + (kt * 32 + kk)];
}

// ---- f32 -> f16 cast with column placement ---------------------------------
__global__ __launch_bounds__(256) void cast_to_f16(const float* __restrict__ src,
                                                   _Float16* __restrict__ dst,
                                                   long n, int srcw, int dstw, int coloff) {
  long e = (long)blockIdx.x * 256 + threadIdx.x;
  if (e >= n) return;
  long r = e / srcw; int c = (int)(e % srcw);
  dst[r * (long)dstw + coloff + c] = (_Float16)src[e];
}

// ---- generic WMMA GEMM: out = A[M,K](f16) @ Bp[K,N] (+bias) (+relu) --------
// grid.x = M/16, grid.y = N/128, 256 threads = 8 waves.
// rowRemap: output row r=(t*16+m) reads A natural row (m*seqT + t).
__global__ __launch_bounds__(256) void gemm_ws(
    const _Float16* __restrict__ A, int lda, int seqT, int rowRemap,
    const _Float16* __restrict__ Bp, int K, int N,
    const float* __restrict__ b1, const float* __restrict__ b2,
    float* __restrict__ outF, _Float16* __restrict__ outH,
    int ldc, int coloff, int relu)
{
  __shared__ _Float16 As[16 * 768];   // A tile [16][K], K<=768 (24KB)
  const int tid = threadIdx.x;
  const int mt = blockIdx.x;
  const int nb = blockIdx.y;
  const int Kt = K >> 5;

  // stage A tile into LDS via async DMA (16B chunks, ASYNCcnt tracked)
  const int chunks = 16 * (K >> 3);
  for (int c = tid; c < chunks; c += 256) {
    int m = c / (K >> 3);
    int kc = c % (K >> 3);
    int ro = mt * 16 + m;
    long ra = rowRemap ? ((long)(ro & 15) * seqT + (ro >> 4)) : (long)ro;
    async_g2l_b128(&As[m * K + kc * 8], A + ra * (long)lda + kc * 8);
  }
  wait_async0();
  __syncthreads();

  const int w = tid >> 5, lane = tid & 31;     // wave32
  const int ntG = nb * 8 + w;
  if (ntG * 16 >= N) return;
  const int n = ntG * 16 + (lane & 15);

  float cinit = 0.f;
  if (b1) cinit += b1[n];
  if (b2) cinit += b2[n];
  v8f c;
  #pragma unroll
  for (int i = 0; i < 8; ++i) c[i] = cinit;

  const _Float16* btile = Bp + ((size_t)ntG * Kt) * 512;
  // register double-buffer; last iteration peeled -> branch-free body.
  v16h a0 = loadA_frag(As, K, lane);
  v16h b0 = loadB_frag(btile, lane);
  #pragma unroll 4
  for (int kt = 0; kt < Kt - 1; ++kt) {
    __builtin_prefetch(btile + (size_t)(kt + 4) * 512, 0, 2);  // speculative OK
    v16h a1 = loadA_frag(As + (kt + 1) * 32, K, lane);
    v16h b1 = loadB_frag(btile + (size_t)(kt + 1) * 512, lane);
    c = wmma_f16(a0, b0, c);
    a0 = a1; b0 = b1;
  }
  c = wmma_f16(a0, b0, c);

  const int hh = lane >> 4;
  #pragma unroll
  for (int v = 0; v < 8; ++v) {
    long r = (long)mt * 16 + v + 8 * hh;       // C/D: M = vgpr + 8*(lane>=16)
    long idx = r * (long)ldc + coloff + n;
    float val = c[v];
    if (relu) val = val > 0.f ? val : 0.f;
    if (outH) outH[idx] = (_Float16)val;
    else      outF[idx] = val;
  }
}

// ---- persistent LSTM scan ---------------------------------------------------
// 24 WGs x 256 threads. WG `blk` owns h columns [blk*32, blk*32+32).
// LDS: w_hh slice (192KB, TDM-loaded) + h bcast (24KB) + gates (8KB) + c (2KB).
#define SCAN_NBLK 24
#define SCAN_LDS  (196608 + 24576 + 8192 + 2048)

__global__ __launch_bounds__(256) void lstm_scan(
    const _Float16* __restrict__ Whp,   // packed w_hh tiles (N=3072, K=768)
    const float*    __restrict__ xg,    // [T][16][3072] (bias included)
    float*          __restrict__ acc,   // [16][T][768] residual accumulator
    _Float16*       __restrict__ inh,   // [16][T][768] f16 next-layer input
    _Float16*       __restrict__ hb,    // [2][16][768] f16 h broadcast
    unsigned* counter, int T, int firstLayer)
{
  extern __shared__ char smem[];
  _Float16* wLDS = (_Float16*)smem;                       // 98304 f16
  _Float16* hLDS = wLDS + 98304;                          // 12288 f16
  float*    gLDS = (float*)(smem + 196608 + 24576);       // [16][128]
  float*    cLDS = gLDS + 2048;                           // [16][32]

  const int tid = threadIdx.x;
  const int blk = blockIdx.x;
  const int w = tid >> 5, lane = tid & 31;

  // ---- preload this WG's w_hh slice: 4 gate-rows x 48KB, row stride 1.125MB.
#if __has_builtin(__builtin_amdgcn_tensor_load_to_lds)
  if (tid < 32) {   // wave 0 issues one 2D TDM descriptor (TENSORcnt tracked)
    uint64_t ga = (uint64_t)(uintptr_t)Whp + (uint64_t)blk * 2 * 24 * 512 * 2;
    u32x4 g0;
    g0[0] = 1u;                                   // count=1 (valid), user mode
    g0[1] = (uint32_t)(uintptr_t)wLDS;            // lds_addr
    g0[2] = (uint32_t)ga;                         // global_addr[31:0]
    g0[3] = (uint32_t)(ga >> 32) | (2u << 30);    // global_addr[56:32], type=2
    i32x8 g1;
    g1[0] = (int)(3u << 16);                      // data_size=3 (8B elements)
    g1[1] = (int)(6144u << 16);                   // tensor_dim0 = 6144
    g1[2] = (int)(4u << 16);                      // tensor_dim1 = 4
    g1[3] = (int)(6144u << 16);                   // tile_dim0 = 6144 (48KB row)
    g1[4] = 4;                                    // tile_dim1 = 4 rows
    g1[5] = 147456;                               // tensor_dim0_stride (8B units)
    g1[6] = 0; g1[7] = 0;
    i32x4 gz = {0, 0, 0, 0};
#if __clang_major__ >= 23
    i32x8 gz8 = {0, 0, 0, 0, 0, 0, 0, 0};
    __builtin_amdgcn_tensor_load_to_lds(g0, g1, gz, gz, gz8, 0);
#else
    __builtin_amdgcn_tensor_load_to_lds(g0, g1, gz, gz, 0);
#endif
    wait_tensor0();
  }
#else
  for (int cidx = tid; cidx < 12288; cidx += 256) {        // fallback copy
    int ww  = cidx / 1536;
    int rem = cidx % 1536;
    int kt = rem / 64, cc = rem % 64;
    int ntG = (ww >> 1) * 48 + blk * 2 + (ww & 1);
    *(v8h*)(wLDS + ((ww * 24 + kt) << 9) + cc * 8) =
        *(const v8h*)(Whp + (((size_t)ntG * 24 + kt) << 9) + cc * 8);
  }
#endif
  for (int i = tid; i < 512; i += 256) cLDS[i] = 0.f;      // c0 = 0

  const int q = w >> 1;                                    // gate (i,f,g,o)
  const int myn = q * 768 + blk * 32 + (w & 1) * 16 + (lane & 15);
  const int hhalf = lane >> 4;

  for (int t = 0; t < T; ++t) {
    if (t > 0) {
      if (tid == 0) {
        const unsigned tgt = (unsigned)t * gridDim.x;
        while (__hip_atomic_load(counter, __ATOMIC_ACQUIRE,
                                 __HIP_MEMORY_SCOPE_AGENT) < tgt)
          __builtin_amdgcn_s_sleep(1);
      }
      __syncthreads();
      // pull h_{t-1} broadcast into LDS (async DMA bypasses WGP$)
      const _Float16* src = hb + (size_t)((t + 1) & 1) * 12288;
      for (int i = tid; i < 1536; i += 256)
        async_g2l_b128(hLDS + i * 8, src + i * 8);
      wait_async0();
    } else {
      for (int i = tid; i < 6144; i += 256) ((unsigned*)hLDS)[i] = 0u;
    }
    __syncthreads();

    // gates = xg[t] + h_{t-1} @ w_hh_slice   (one 16x16 tile per wave)
    v8f cfrag;
    {
      const float* xr = xg + (size_t)t * 16 * 3072 + myn;
      #pragma unroll
      for (int v = 0; v < 8; ++v)
        cfrag[v] = xr[(size_t)(v + 8 * hhalf) * 3072];
      // hide next step's xg HBM latency behind this step's WMMA chain
      __builtin_prefetch(xr + (size_t)16 * 3072, 0, 2);
      __builtin_prefetch(xr + (size_t)(16 + 8) * 3072, 0, 2);
    }
    const _Float16* wt = wLDS + (size_t)(w * 24) * 512;
    v16h a0 = loadA_frag(hLDS, 768, lane);
    v16h b0 = loadB_frag(wt, lane);
    #pragma unroll 4
    for (int kt = 0; kt < 23; ++kt) {
      v16h a1 = loadA_frag(hLDS + (kt + 1) * 32, 768, lane);
      v16h b1 = loadB_frag(wt + (size_t)(kt + 1) * 512, lane);
      cfrag = wmma_f16(a0, b0, cfrag);
      a0 = a1; b0 = b1;
    }
    cfrag = wmma_f16(a0, b0, cfrag);
    {
      const int lc = w * 16 + (lane & 15);
      #pragma unroll
      for (int v = 0; v < 8; ++v)
        gLDS[(v + 8 * hhalf) * 128 + lc] = cfrag[v];
    }
    __syncthreads();

    // elementwise LSTM cell for 16 batch x 32 owned h-columns
    for (int e = tid; e < 512; e += 256) {
      int m = e >> 5, j = e & 31;
      float gi = gLDS[m * 128 + j];
      float gf = gLDS[m * 128 + 32 + j];
      float gg = gLDS[m * 128 + 64 + j];
      float go = gLDS[m * 128 + 96 + j];
      float cprev = cLDS[e];
      float si = 1.f / (1.f + __expf(-gi));
      float sf = 1.f / (1.f + __expf(-gf));
      float so = 1.f / (1.f + __expf(-go));
      float tg = 1.f - 2.f / (__expf(2.f * gg) + 1.f);
      float cn = sf * cprev + si * tg;
      cLDS[e] = cn;
      float tc = 1.f - 2.f / (__expf(2.f * cn) + 1.f);
      float h  = so * tc;
      int col = blk * 32 + j;
      hb[(size_t)(t & 1) * 12288 + m * 768 + col] = (_Float16)h;
      size_t ridx = ((size_t)m * T + t) * 768 + col;
      float an = firstLayer ? h : (acc[ridx] + h);
      acc[ridx] = an;
      inh[ridx] = (_Float16)an;
    }
    __threadfence();
    __syncthreads();
    if (tid == 0)
      __hip_atomic_fetch_add(counter, 1u, __ATOMIC_RELEASE,
                             __HIP_MEMORY_SCOPE_AGENT);
  }
}

// ---------------------------------------------------------------------------
extern "C" void kernel_launch(void* const* d_in, const int* in_sizes, int n_in,
                              void* d_out, int out_size, void* d_ws, size_t ws_size,
                              hipStream_t stream) {
  (void)in_sizes; (void)n_in; (void)out_size; (void)ws_size;
  const float* x    = (const float*)d_in[0];
  const float* mels = (const float*)d_in[1];
  const float* pw1  = (const float*)d_in[2];
  const float* pb1  = (const float*)d_in[3];
  const float* pw2  = (const float*)d_in[4];
  const float* pb2  = (const float*)d_in[5];
  const float* w_ih[3] = {(const float*)d_in[6],  (const float*)d_in[10], (const float*)d_in[14]};
  const float* w_hh[3] = {(const float*)d_in[7],  (const float*)d_in[11], (const float*)d_in[15]};
  const float* b_ih[3] = {(const float*)d_in[8],  (const float*)d_in[12], (const float*)d_in[16]};
  const float* b_hh[3] = {(const float*)d_in[9],  (const float*)d_in[13], (const float*)d_in[17]};
  const float* proj = (const float*)d_in[18];

  const int B = 16, T = 1000, M = B * T;   // 16000 rows

  char* p = (char*)d_ws;
  auto carve = [&](size_t bytes) -> char* {
    char* r = p; p += (bytes + 255) & ~(size_t)255; return r;
  };
  _Float16* in_f16 = (_Float16*)carve((size_t)M * 768 * 2);
  _Float16* mels16 = (_Float16*)carve((size_t)M * 128 * 2);
  _Float16* m1_16  = (_Float16*)carve((size_t)M * 256 * 2);
  float*    xg     = (float*)  carve((size_t)M * 3072 * 4);   // [T][B][4H]
  float*    acc    = (float*)  carve((size_t)M * 768 * 4);
  _Float16* hb     = (_Float16*)carve(2 * 16 * 768 * 2);
  _Float16* pw1p   = (_Float16*)carve((size_t)256 * 128 * 2);
  _Float16* pw2p   = (_Float16*)carve((size_t)256 * 256 * 2);
  _Float16 *wihp[3], *whhp[3];
  for (int i = 0; i < 3; ++i) {
    wihp[i] = (_Float16*)carve((size_t)3072 * 768 * 2);
    whhp[i] = (_Float16*)carve((size_t)3072 * 768 * 2);
  }
  _Float16* projp    = (_Float16*)carve((size_t)128 * 768 * 2);
  unsigned* counters = (unsigned*)carve(256);

  hipMemsetAsync(counters, 0, 256, stream);

  auto packL = [&](const float* W, _Float16* o, int N, int K) {
    size_t tot = (size_t)N * K;
    pack_w<<<dim3((unsigned)((tot + 255) / 256)), 256, 0, stream>>>(W, o, N, K);
  };
  packL(pw1, pw1p, 256, 128);
  packL(pw2, pw2p, 256, 256);
  for (int i = 0; i < 3; ++i) { packL(w_ih[i], wihp[i], 3072, 768); packL(w_hh[i], whhp[i], 3072, 768); }
  packL(proj, projp, 128, 768);

  { long n = (long)M * 512;
    cast_to_f16<<<dim3((unsigned)((n + 255) / 256)), 256, 0, stream>>>(x, in_f16, n, 512, 768, 0); }
  { long n = (long)M * 128;
    cast_to_f16<<<dim3((unsigned)((n + 255) / 256)), 256, 0, stream>>>(mels, mels16, n, 128, 128, 0); }

  // PreNet: relu(mels@pw1.T+pb1) -> m1 ; relu(m1@pw2.T+pb2) -> in_f16[:,512:768]
  gemm_ws<<<dim3(M / 16, 2), 256, 0, stream>>>(mels16, 128, T, 0, pw1p, 128, 256,
                                               pb1, nullptr, nullptr, m1_16, 256, 0, 1);
  gemm_ws<<<dim3(M / 16, 2), 256, 0, stream>>>(m1_16, 256, T, 0, pw2p, 256, 256,
                                               pb2, nullptr, nullptr, in_f16, 768, 512, 1);

  hipFuncSetAttribute(reinterpret_cast<const void*>(lstm_scan),
                      hipFuncAttributeMaxDynamicSharedMemorySize, SCAN_LDS);

  for (int L = 0; L < 3; ++L) {
    // xg[T][B][4H] = input @ w_ih.T + b_ih + b_hh   (rowRemap: out row t*16+m)
    gemm_ws<<<dim3(M / 16, 24), 256, 0, stream>>>(in_f16, 768, T, 1, wihp[L], 768, 3072,
                                                  b_ih[L], b_hh[L], xg, nullptr, 3072, 0, 0);
    lstm_scan<<<dim3(SCAN_NBLK), 256, SCAN_LDS, stream>>>(whhp[L], xg, acc, in_f16, hb,
                                                          counters + 16 * L, T, L == 0);
  }

  // final projection: a3 @ proj_w.T -> d_out [B,T,128] (f32)
  gemm_ws<<<dim3(M / 16, 1), 256, 0, stream>>>(in_f16, 768, T, 0, projp, 768, 128,
                                               nullptr, nullptr, (float*)d_out, nullptr, 128, 0, 0);
}
